// GaussianVectorQuantizer_61787399520297
// MI455X (gfx1250) — matrix operand order repaired and checked by reading, not compiled, not run on hardware
//
#include <hip/hip_runtime.h>

// MI455X / gfx1250, wave32. Fused Gaussian VQ forward.
// Roofline: ~288MB compulsory traffic / 23.3 TB/s ~= 12us floor. Matrix work
// uses V_WMMA_F32_16X16X32_F16 (f16 in, f32 acc). The codebook (256KB,
// L2-resident) is pre-converted once to f16 in both row-major and transposed
// layouts so every WMMA B operand is a contiguous 32B load, and all
// per-chunk codebook conversion/square/reduction VALU+LDS work disappears.

typedef __attribute__((ext_vector_type(16))) _Float16 v16h;
typedef __attribute__((ext_vector_type(8)))  _Float16 v8h;
typedef __attribute__((ext_vector_type(8)))  float    v8f;

#define BS   65536
#define DIMD 64
#define SD   1024
#define WAVES_PER_WG 8

// workspace layout (bytes):
//   [0      .. 4096 )  avg_prob sums   (1024 f32)
//   [4096   .. 4100 )  loss accumulator
//   [4352   .. 8448 )  cnorm           (1024 f32)
//   [8448   .. 139520) Ch  f16 codebook row-major   [1024][64]
//   [139520 .. 270592) Ct  f16 codebook transposed  [64][1024]
#define WS_CNORM_F  1088
#define WS_CH_BYTE  8448
#define WS_CT_BYTE  139520
#define WS_NEEDED   270592

__device__ __forceinline__ float wave16_max(float x) {
#pragma unroll
  for (int m = 1; m < 16; m <<= 1) x = fmaxf(x, __shfl_xor(x, m, 32));
  return x;
}
__device__ __forceinline__ float wave16_sum(float x) {
#pragma unroll
  for (int m = 1; m < 16; m <<= 1) x += __shfl_xor(x, m, 32);
  return x;
}

__global__ void gvq_zero(float* ws) {
  int i = blockIdx.x * blockDim.x + threadIdx.x;
  if (i < 1025) ws[i] = 0.0f;
}

// One-time codebook prep: f16 row-major, f16 transposed, row norms.
__global__ void gvq_prep(const float* __restrict__ C, _Float16* __restrict__ Ch,
                         _Float16* __restrict__ Ct, float* __restrict__ CN) {
  int j = blockIdx.x * blockDim.x + threadIdx.x;
  if (j < SD) {
    float s = 0.f;
#pragma unroll
    for (int d = 0; d < DIMD; ++d) {
      float v = C[(size_t)j * DIMD + d];
      s += v * v;
      _Float16 h = (_Float16)v;
      Ch[(size_t)j * DIMD + d] = h;
      Ct[(size_t)d * SD + j]   = h;
    }
    CN[j] = s;
  }
}

template <bool F16CB>
__global__ __launch_bounds__(256) void gvq_main(
    const float* __restrict__ Z, const float* __restrict__ PQ,
    const float* __restrict__ C, const float* __restrict__ U,
    float* __restrict__ zq_out, float* __restrict__ ws,
    const _Float16* __restrict__ Ch, const _Float16* __restrict__ Ct,
    const float* __restrict__ CN) {
  __shared__ __align__(16) _Float16 e_s[WAVES_PER_WG][16][32];
  __shared__ float zn_s[WAVES_PER_WG][16];

  const int lane = threadIdx.x & 31;
  const int wave = threadIdx.x >> 5;
  const int half = lane >> 4;
  const int l16  = lane & 15;
  const int row0 = (blockIdx.x * WAVES_PER_WG + wave) * 16;

  const float w = 0.5f / fmaxf(PQ[0], 1e-10f);

  // ----- WMMA A operands (Z tile f16) + znorm ------------------------------
  // A layout (16x32 f16): lane<16 row=lane K0..15;
  // lane>=16 row=lane-16 K16..31.
  v16h A0, A1;
  float sq = 0.0f;
  {
    const float* zr = Z + (size_t)(row0 + l16) * DIMD + half * 16;
#pragma unroll
    for (int g = 0; g < 4; ++g) {
      const float* p = zr + (g >> 1) * 32 + (g & 1) * 8;
      float4 fa = *(const float4*)(p);
      float4 fb = *(const float4*)(p + 4);
      float f[8] = {fa.x, fa.y, fa.z, fa.w, fb.x, fb.y, fb.z, fb.w};
#pragma unroll
      for (int i = 0; i < 8; ++i) {
        sq += f[i] * f[i];
        _Float16 h = (_Float16)f[i];
        if (g == 0)      A0[i]     = h;
        else if (g == 1) A0[8 + i] = h;
        else if (g == 2) A1[i]     = h;
        else             A1[8 + i] = h;
      }
    }
  }
  sq += __shfl_xor(sq, 16, 32);
  if (half == 0) zn_s[wave][l16] = sq;
  __syncthreads();

  float zn_e[8];
#pragma unroll
  for (int e = 0; e < 8; ++e) zn_e[e] = zn_s[wave][e + half * 8];

  // ----- one 16-code column tile of logits ---------------------------------
  // B layout (32x16 f16): lane<16 col=lane halves=K0..15; lane>=16 K16..31
  // -> contiguous 32B per lane from the f16 codebook row.
  auto logits_tile = [&](int n0, float* lt) {
    const int n = n0 + l16;
    v16h B0, B1;
    float csq;
    if constexpr (F16CB) {
      B0  = *(const v16h*)(Ch + (size_t)n * DIMD + half * 16);
      B1  = *(const v16h*)(Ch + (size_t)n * DIMD + 32 + half * 16);
      csq = CN[n];
    } else {
      const float* cr = C + (size_t)n * DIMD + half * 16;
      csq = 0.0f;
#pragma unroll
      for (int kc = 0; kc < 2; ++kc) {
        const float* p = cr + kc * 32;
        float4 fa = *(const float4*)(p);
        float4 fb = *(const float4*)(p + 4);
        float4 fc = *(const float4*)(p + 8);
        float4 fd = *(const float4*)(p + 12);
        float f[16] = {fa.x, fa.y, fa.z, fa.w, fb.x, fb.y, fb.z, fb.w,
                       fc.x, fc.y, fc.z, fc.w, fd.x, fd.y, fd.z, fd.w};
#pragma unroll
        for (int i = 0; i < 16; ++i) {
          csq += f[i] * f[i];
          if (kc == 0) B0[i] = (_Float16)f[i];
          else         B1[i] = (_Float16)f[i];
        }
      }
      csq += __shfl_xor(csq, 16, 32);
    }
    v8f acc = {};
    acc = __builtin_amdgcn_wmma_f32_16x16x32_f16(false, A0, false, B0,
                                                 (short)0, acc, false, false);
    acc = __builtin_amdgcn_wmma_f32_16x16x32_f16(false, A1, false, B1,
                                                 (short)0, acc, false, false);
#pragma unroll
    for (int e = 0; e < 8; ++e)
      lt[e] = w * (2.0f * acc[e] - zn_e[e] - csq);   // = -w * dist
  };

  // ----- Pass 1: online row max + normalizer (64 codes / iteration) --------
  float mp[8], Ap[8];
#pragma unroll
  for (int e = 0; e < 8; ++e) { mp[e] = -1e30f; Ap[e] = 0.f; }

  for (int c = 0; c < 16; ++c) {
    const int j0 = c * 64;
    float lt[4][8];
#pragma unroll
    for (int t = 0; t < 4; ++t) logits_tile(j0 + t * 16, lt[t]);
    float cm[8], sarr[8], ra[8], nm[8];
#pragma unroll
    for (int e = 0; e < 8; ++e) {
      float m = fmaxf(fmaxf(lt[0][e], lt[1][e]), fmaxf(lt[2][e], lt[3][e]));
      cm[e] = wave16_max(m);
    }
#pragma unroll
    for (int e = 0; e < 8; ++e) {
      nm[e]   = fmaxf(mp[e], cm[e]);
      sarr[e] = __expf(mp[e] - nm[e]);
      ra[e]   = __expf(lt[0][e] - nm[e]) + __expf(lt[1][e] - nm[e]) +
                __expf(lt[2][e] - nm[e]) + __expf(lt[3][e] - nm[e]);
    }
#pragma unroll
    for (int e = 0; e < 8; ++e) ra[e] = wave16_sum(ra[e]);
#pragma unroll
    for (int e = 0; e < 8; ++e) { Ap[e] = Ap[e] * sarr[e] + ra[e]; mp[e] = nm[e]; }
  }

  float invAp[8], kconst[8];
#pragma unroll
  for (int e = 0; e < 8; ++e) {
    invAp[e]  = 1.0f / Ap[e];
    kconst[e] = mp[e] + __logf(Ap[e]);   // log-sum-exp; log p = l - kconst
  }

  // ----- Pass 2: avg_probs + kld_discrete + gumbel flash softmax + GEMM2 ---
  float mg[8], Sg[8], kd_acc = 0.f;
  v8f ZQ[4];
#pragma unroll
  for (int e = 0; e < 8; ++e) { mg[e] = -1e30f; Sg[e] = 0.f; }
#pragma unroll
  for (int t = 0; t < 4; ++t) {
#pragma unroll
    for (int e = 0; e < 8; ++e) ZQ[t][e] = 0.f;
  }
  v16h Bones;
#pragma unroll
  for (int i = 0; i < 16; ++i) Bones[i] = (_Float16)1.0f;

  for (int c = 0; c < 32; ++c) {
    const int j0 = c * 32;
    float l0[8], l1[8];
    logits_tile(j0, l0);
    logits_tile(j0 + 16, l1);

    // probabilities: column sums -> global atomics; p*(log p) -> lane partial
    float ps0 = 0.f, ps1 = 0.f;
#pragma unroll
    for (int e = 0; e < 8; ++e) {
      float p0 = __expf(l0[e] - mp[e]) * invAp[e];
      float p1 = __expf(l1[e] - mp[e]) * invAp[e];
      ps0 += p0;
      ps1 += p1;
      kd_acc += p0 * (l0[e] - kconst[e]) + p1 * (l1[e] - kconst[e]);
    }
    ps0 += __shfl_xor(ps0, 16, 32);
    ps1 += __shfl_xor(ps1, 16, 32);
    if (half == 0) {
      unsafeAtomicAdd(&ws[j0 + l16], ps0);
      unsafeAtomicAdd(&ws[j0 + 16 + l16], ps1);
    }

    // gumbel perturbation (U streamed exactly once)
    float y0[8], y1[8];
#pragma unroll
    for (int e = 0; e < 8; ++e) {
      const size_t ub = (size_t)(row0 + e + half * 8) * SD + j0 + l16;
      float u0 = U[ub], u1 = U[ub + 16];
      if (e == 0 && c + 1 < 32)
        __builtin_prefetch((const void*)(U + ub + 32), 0, 0);
      float g0 = -__logf(-__logf(u0 + 1e-10f) + 1e-10f);
      float g1 = -__logf(-__logf(u1 + 1e-10f) + 1e-10f);
      y0[e] = l0[e] + g0;
      y1[e] = l1[e] + g1;
    }
    float cm[8], sarr[8], eg0[8], eg1[8];
#pragma unroll
    for (int e = 0; e < 8; ++e) cm[e] = wave16_max(fmaxf(y0[e], y1[e]));
#pragma unroll
    for (int e = 0; e < 8; ++e) {
      float nm = fmaxf(mg[e], cm[e]);
      sarr[e] = __expf(mg[e] - nm);
      eg0[e]  = __expf(y0[e] - nm);
      eg1[e]  = __expf(y1[e] - nm);
      mg[e]   = nm;
    }

    // D-layout -> WMMA A-layout transpose of e-tile through LDS scratch
#pragma unroll
    for (int e = 0; e < 8; ++e) {
      const int r = e + half * 8;
      e_s[wave][r][l16]      = (_Float16)eg0[e];
      e_s[wave][r][l16 + 16] = (_Float16)eg1[e];
    }
    __threadfence_block();
    v8h lo = *(const v8h*)&e_s[wave][l16][half * 16];
    v8h hi = *(const v8h*)&e_s[wave][l16][half * 16 + 8];
    v16h Ae;
#pragma unroll
    for (int i = 0; i < 8; ++i) { Ae[i] = lo[i]; Ae[8 + i] = hi[i]; }

    // flash denominator via WMMA against ones: D[m][*] = sum_k e[m][k]
    // (same f16-rounded values as the numerator -> self-consistent)
    {
      v8f rs = {};
      rs = __builtin_amdgcn_wmma_f32_16x16x32_f16(false, Ae, false, Bones,
                                                  (short)0, rs, false, false);
#pragma unroll
      for (int e = 0; e < 8; ++e) Sg[e] = Sg[e] * sarr[e] + rs[e];
    }
#pragma unroll
    for (int t = 0; t < 4; ++t) {
#pragma unroll
      for (int e = 0; e < 8; ++e) ZQ[t][e] *= sarr[e];
    }

    // GEMM2: zq += e_tile (16x32) @ C[j0..j0+31, :]
#pragma unroll
    for (int dt = 0; dt < 4; ++dt) {
      v16h B2;
      if constexpr (F16CB) {
        B2 = *(const v16h*)(Ct + (size_t)(dt * 16 + l16) * SD + j0 + half * 16);
      } else {
        const float* cp = C + (size_t)(j0 + half * 16) * DIMD + dt * 16 + l16;
#pragma unroll
        for (int k = 0; k < 16; ++k) B2[k] = (_Float16)cp[(size_t)k * DIMD];
      }
      ZQ[dt] = __builtin_amdgcn_wmma_f32_16x16x32_f16(false, Ae, false, B2,
                                                      (short)0, ZQ[dt], false, false);
    }
  }

  // ----- finalize: zq = num/den, store, kld_continuous ---------------------
  float inv[8];
#pragma unroll
  for (int e = 0; e < 8; ++e) inv[e] = 1.0f / Sg[e];
  float kc_acc = 0.f;
#pragma unroll
  for (int dt = 0; dt < 4; ++dt) {
#pragma unroll
    for (int e = 0; e < 8; ++e) {
      const int row = row0 + e + half * 8;
      const int dim = dt * 16 + l16;
      float q = ZQ[dt][e] * inv[e];
      zq_out[(size_t)row * DIMD + dim] = q;
      float d = Z[(size_t)row * DIMD + dim] - q;
      kc_acc += d * d;
    }
  }
  float acc = kd_acc + w * kc_acc;      // both terms are /bs later
#pragma unroll
  for (int m = 1; m < 32; m <<= 1) acc += __shfl_xor(acc, m, 32);
  if (lane == 0) unsafeAtomicAdd(&ws[1024], acc);
}

__global__ __launch_bounds__(1024) void gvq_fin(const float* __restrict__ ws,
                                                float* __restrict__ out) {
  __shared__ float red[1024];
  const int t = threadIdx.x;
  float a = ws[t] * (1.0f / (float)BS);
  red[t] = a * __logf(a + 1e-7f);
  __syncthreads();
  for (int s = 512; s > 0; s >>= 1) {
    if (t < s) red[t] += red[t + s];
    __syncthreads();
  }
  if (t == 0) {
    out[(size_t)BS * DIMD]     = ws[1024] * (1.0f / (float)BS);  // loss
    out[(size_t)BS * DIMD + 1] = __expf(-red[0]);                // perplexity
  }
}

extern "C" void kernel_launch(void* const* d_in, const int* in_sizes, int n_in,
                              void* d_out, int out_size, void* d_ws, size_t ws_size,
                              hipStream_t stream) {
  const float* Z  = (const float*)d_in[0];
  const float* PQ = (const float*)d_in[1];
  const float* C  = (const float*)d_in[2];
  const float* U  = (const float*)d_in[3];
  float* out = (float*)d_out;
  float* ws  = (float*)d_ws;

  gvq_zero<<<5, 256, 0, stream>>>(ws);

  const int grid = BS / (16 * WAVES_PER_WG);
  if (ws_size >= (size_t)WS_NEEDED) {
    _Float16* Ch = (_Float16*)((char*)d_ws + WS_CH_BYTE);
    _Float16* Ct = (_Float16*)((char*)d_ws + WS_CT_BYTE);
    float*    CN = ws + WS_CNORM_F;
    gvq_prep<<<4, 256, 0, stream>>>(C, Ch, Ct, CN);
    gvq_main<true><<<grid, 32 * WAVES_PER_WG, 0, stream>>>(
        Z, PQ, C, U, out, ws, Ch, Ct, CN);
  } else {
    gvq_main<false><<<grid, 32 * WAVES_PER_WG, 0, stream>>>(
        Z, PQ, C, U, out, ws, nullptr, nullptr, nullptr);
  }
  gvq_fin<<<1, 1024, 0, stream>>>(ws, out);
}
